// XPredNextScale_11862699671587
// MI455X (gfx1250) — matrix operand order repaired and checked by reading, not compiled, not run on hardware
//
#include <hip/hip_runtime.h>
#include <hip/hip_bf16.h>
#include <math.h>

#define DEV __device__ __forceinline__

typedef __bf16 bf16_t;
typedef bf16_t v16bf __attribute__((ext_vector_type(16)));
typedef float  v8f   __attribute__((ext_vector_type(8)));

static constexpr int Dm  = 768;
static constexpr int Lm  = 341;
static constexpr int Bm  = 16;
static constexpr int SLD = 352;   // scores leading dim (padded 341 -> 352)
#define LDST 40                    // LDS row stride in ushorts (80B, 16B aligned)

#define GF_BIAS  1
#define GF_GELU  2
#define GF_RES   4
#define GF_REMAP 8

DEV unsigned short f2bf(float f){
  unsigned u = __float_as_uint(f);
  u += 0x7FFFu + ((u >> 16) & 1u);        // round-to-nearest-even
  return (unsigned short)(u >> 16);
}
DEV unsigned pack_bf2(float a, float b){
  return (unsigned)f2bf(a) | ((unsigned)f2bf(b) << 16);
}

union FragU { uint4 q[2]; v16bf v; };

// A fragment: 16x32 (MxK), lane m = row, half h: K = {h*8..h*8+7} U {16+h*8..16+h*8+7}
DEV v16bf load_a_frag(const unsigned short* s, int mrow, int lane){
  int m = mrow + (lane & 15), h = (lane >> 4) & 1;
  const unsigned short* p = s + m * LDST + h * 8;
  FragU f; f.q[0] = *(const uint4*)p; f.q[1] = *(const uint4*)(p + 16);
  return f.v;
}
// B fragment: 32x16 (KxN) stored transposed in LDS as Bt[N][K]; lane n = col, half h: K = {h*8..h*8+7} U {16+h*8..16+h*8+7}
DEV v16bf load_b_frag(const unsigned short* s, int nrow, int lane){
  int n = nrow + (lane & 15), h = (lane >> 4) & 1;
  const unsigned short* p = s + n * LDST + h * 8;
  FragU f; f.q[0] = *(const uint4*)p; f.q[1] = *(const uint4*)(p + 16);
  return f.v;
}

// guarded float4 load (16B-aligned fast path, per-element tail)
DEV float4 ld4_guard(const float* p, long long row_off, int gk, int K, bool row_ok){
  float4 v = make_float4(0.f, 0.f, 0.f, 0.f);
  if (row_ok){
    if (gk + 3 < K){
      v = *(const float4*)(p + row_off + gk);
    } else {
      if (gk + 0 < K) v.x = p[row_off + gk + 0];
      if (gk + 1 < K) v.y = p[row_off + gk + 1];
      if (gk + 2 < K) v.z = p[row_off + gk + 2];
      if (gk + 3 < K) v.w = p[row_off + gk + 3];
    }
  }
  return v;
}

// ---------------- Generic batched WMMA GEMM ----------------
// Block = 128 threads (4 waves). Tile BM=128, BN=64, BK=32.
// Each wave owns a 128x16 output strip: 8 accumulator tiles -> 8 WMMAs per K-step.
__global__ __launch_bounds__(128)
void gemm_wmma(const float* __restrict__ A, int lda, long long a_sb, long long a_sh,
               const float* __restrict__ Bsrc, int ldb, long long b_sb, long long b_sh, int b_is_nk,
               float* __restrict__ C, int ldc, long long c_sb, long long c_sh,
               const float* __restrict__ bias, const float* __restrict__ resid,
               int M, int N, int K, int zh, int flags)
{
  __shared__ __align__(16) unsigned short As[128 * LDST];
  __shared__ __align__(16) unsigned short Bs[64 * LDST];
  int bn = blockIdx.x * 64, bm = blockIdx.y * 128;
  int z = blockIdx.z, zb = z / zh, zi = z % zh;
  const float* Ab = A    + (long long)zb * a_sb + (long long)zi * a_sh;
  const float* Bb = Bsrc + (long long)zb * b_sb + (long long)zi * b_sh;
  long long coff = (long long)zb * c_sb + (long long)zi * c_sh;
  int tid = threadIdx.x, lane = tid & 31, wave = tid >> 5;

  v8f acc[8] = {};

  for (int k0 = 0; k0 < K; k0 += 32){
    // prefetch next K slice of A into near caches
    if (k0 + 32 < K){
      int pr = bm + tid;
      __builtin_prefetch(Ab + (long long)pr * lda + k0 + 32, 0, 3);
    }
    // ---- stage A tile 128x32 as bf16 (vec4 global load, b64 LDS store) ----
#pragma unroll
    for (int it = 0; it < 8; ++it){
      int i = tid + it * 128;                 // 1024 float4 chunks
      int r = i >> 3, c4 = (i & 7) * 4;
      int gm = bm + r, gk = k0 + c4;
      float4 v = ld4_guard(Ab, (long long)gm * lda, gk, K, gm < M);
      uint2 p; p.x = pack_bf2(v.x, v.y); p.y = pack_bf2(v.z, v.w);
      *(uint2*)(As + r * LDST + c4) = p;
    }
    // ---- stage B tile into Bt[N=64][K=32] ----
    if (b_is_nk){
#pragma unroll
      for (int it = 0; it < 4; ++it){
        int i = tid + it * 128;               // 512 float4 chunks
        int n = i >> 3, c4 = (i & 7) * 4;
        int gn = bn + n, gk = k0 + c4;
        float4 v = ld4_guard(Bb, (long long)gn * ldb, gk, K, gn < N);
        uint2 p; p.x = pack_bf2(v.x, v.y); p.y = pack_bf2(v.z, v.w);
        *(uint2*)(Bs + n * LDST + c4) = p;
      }
    } else {
      // B row-major [K,N]: vec4 along n, transpose into Bt via 4 b16 stores
#pragma unroll
      for (int it = 0; it < 4; ++it){
        int i = tid + it * 128;               // 512 float4 chunks
        int kk = i >> 4, n4 = (i & 15) * 4;
        int gn = bn + n4, gk = k0 + kk;
        float4 v = make_float4(0.f, 0.f, 0.f, 0.f);
        bool kok = gk < K;
        if (kok){
          if (gn + 3 < N){
            v = *(const float4*)(Bb + (long long)gk * ldb + gn);
          } else {
            if (gn + 0 < N) v.x = Bb[(long long)gk * ldb + gn + 0];
            if (gn + 1 < N) v.y = Bb[(long long)gk * ldb + gn + 1];
            if (gn + 2 < N) v.z = Bb[(long long)gk * ldb + gn + 2];
            if (gn + 3 < N) v.w = Bb[(long long)gk * ldb + gn + 3];
          }
        }
        Bs[(n4 + 0) * LDST + kk] = f2bf(v.x);
        Bs[(n4 + 1) * LDST + kk] = f2bf(v.y);
        Bs[(n4 + 2) * LDST + kk] = f2bf(v.z);
        Bs[(n4 + 3) * LDST + kk] = f2bf(v.w);
      }
    }
    __syncthreads();
    v16bf bf = load_b_frag(Bs, wave * 16, lane);
#pragma unroll
    for (int mt = 0; mt < 8; ++mt){
      v16bf af = load_a_frag(As, mt * 16, lane);
      acc[mt] = __builtin_amdgcn_wmma_f32_16x16x32_bf16(false, af, false, bf,
                                                        (short)0, acc[mt], false, false);
    }
    __syncthreads();
  }

  // epilogue: D layout — VGPR r: M=r (lanes 0-15) / M=r+8 (lanes 16-31), N=lane&15
  int half = (lane >> 4) & 1, nl = lane & 15;
  int gn = bn + wave * 16 + nl;
  if (gn < N){
    float bv = (flags & GF_BIAS) ? bias[gn] : 0.f;
#pragma unroll
    for (int mt = 0; mt < 8; ++mt){
#pragma unroll
      for (int r = 0; r < 8; ++r){
        int gm = bm + mt * 16 + r + half * 8;
        if (gm < M){
          float f = acc[mt][r] + bv;
          if (flags & GF_GELU){
            float u = 0.7978845608028654f * (f + 0.044715f * f * f * f);
            f = 0.5f * f * (1.f + tanhf(u));
          }
          long long orow = (flags & GF_REMAP) ? ((long long)(gm / 340) * 341 + 1 + gm % 340)
                                              : (long long)gm;
          long long ci = coff + orow * (long long)ldc + gn;
          if (flags & GF_RES) f += resid[ci];
          C[ci] = f;
        }
      }
    }
  }
}

// ---------------- LayerNorm (one block per row of 768) ----------------
__global__ __launch_bounds__(256)
void layernorm_k(const float* __restrict__ X, const float* __restrict__ g,
                 const float* __restrict__ b, float* __restrict__ O)
{
  __shared__ float red[256];
  long long r = blockIdx.x;
  const float* x = X + r * Dm;
  float* o = O + r * Dm;
  int t = threadIdx.x;
  float v0 = x[t], v1 = x[t + 256], v2 = x[t + 512];
  red[t] = v0 + v1 + v2; __syncthreads();
  for (int s = 128; s > 0; s >>= 1){ if (t < s) red[t] += red[t + s]; __syncthreads(); }
  float mu = red[0] * (1.f / 768.f);
  __syncthreads();
  float d0 = v0 - mu, d1 = v1 - mu, d2 = v2 - mu;
  red[t] = d0 * d0 + d1 * d1 + d2 * d2; __syncthreads();
  for (int s = 128; s > 0; s >>= 1){ if (t < s) red[t] += red[t + s]; __syncthreads(); }
  float rs = rsqrtf(red[0] * (1.f / 768.f) + 1e-5f);
  o[t]       = d0 * rs * g[t]       + b[t];
  o[t + 256] = d1 * rs * g[t + 256] + b[t + 256];
  o[t + 512] = d2 * rs * g[t + 512] + b[t + 512];
}

// ---------------- block-causal masked softmax ----------------
DEV int sid_of(int t){ return (t < 1) ? 0 : (t < 5) ? 1 : (t < 21) ? 2 : (t < 85) ? 3 : 4; }

__global__ __launch_bounds__(128)
void softmax_mask_k(float* __restrict__ S)
{
  __shared__ float red[128];
  long long row = blockIdx.x;          // b*NH*L + h*L + i
  int i = (int)(row % Lm);
  float* p = S + row * SLD;
  int si = sid_of(i);
  int t = threadIdx.x;
  const float scl = 0.125f;            // 1/sqrt(64)
  float mx = -3.4e38f;
  for (int j = t; j < Lm; j += 128){
    float v = (sid_of(j) <= si) ? p[j] * scl : -1e9f;
    mx = fmaxf(mx, v);
  }
  red[t] = mx; __syncthreads();
  for (int s = 64; s > 0; s >>= 1){ if (t < s) red[t] = fmaxf(red[t], red[t + s]); __syncthreads(); }
  mx = red[0]; __syncthreads();
  float sm = 0.f;
  for (int j = t; j < Lm; j += 128){
    float v = (sid_of(j) <= si) ? __expf(p[j] * scl - mx) : 0.f;
    p[j] = v; sm += v;
  }
  red[t] = sm; __syncthreads();
  for (int s = 64; s > 0; s >>= 1){ if (t < s) red[t] += red[t + s]; __syncthreads(); }
  float inv = 1.f / red[0];
  for (int j = t; j < Lm; j += 128) p[j] *= inv;
}

// ---------------- 2x2 mean pool cascade ----------------
__global__ void pool2x2_k(const float* __restrict__ in, float* __restrict__ out, int n, int ho)
{
  int idx = blockIdx.x * 256 + threadIdx.x; if (idx >= n) return;
  int x = idx % ho; int y = (idx / ho) % ho; int c = idx / (ho * ho);
  int hi = ho * 2;
  const float* s = in + (long long)c * hi * hi;
  float v = s[(2*y)*hi + 2*x] + s[(2*y)*hi + 2*x + 1] + s[(2*y+1)*hi + 2*x] + s[(2*y+1)*hi + 2*x + 1];
  out[idx] = 0.25f * v;
}

DEV void tok_decode(int t, int& k, int& g, int& j){
  if (t >= 85)      { k = 4; g = 16; j = t - 85; }
  else if (t >= 21) { k = 3; g = 8;  j = t - 21; }
  else if (t >= 5)  { k = 2; g = 4;  j = t - 5; }
  else if (t >= 1)  { k = 1; g = 2;  j = t - 1; }
  else              { k = 0; g = 1;  j = 0; }
}

// ---------------- patchified targets ----------------
__global__ void targets_k(const float* __restrict__ imgs, const float* __restrict__ P16,
                          const float* __restrict__ P32, const float* __restrict__ P64,
                          const float* __restrict__ P128, float* __restrict__ T)
{
  long long idx = blockIdx.x * 256ll + threadIdx.x;
  long long total = (long long)Bm * Lm * Dm;
  if (idx >= total) return;
  int e = (int)(idx % Dm); int t = (int)((idx / Dm) % Lm); int b = (int)(idx / ((long long)Dm * Lm));
  int k, g, j; tok_decode(t, k, g, j);
  int gy = j / g, gx = j % g;
  int c = e >> 8, py = (e >> 4) & 15, px = e & 15;
  const float* src; int s;
  switch (k){ case 0: src = P16;  s = 16;  break;
              case 1: src = P32;  s = 32;  break;
              case 2: src = P64;  s = 64;  break;
              case 3: src = P128; s = 128; break;
              default: src = imgs; s = 256; }
  long long off = (((long long)b * 3 + c) * s + (gy * 16 + py)) * s + gx * 16 + px;
  T[idx] = src[off];
}

// ---------------- bicubic x2 upsample fused with patchify ----------------
__global__ void upp_k(const float* __restrict__ P16, const float* __restrict__ P32,
                      const float* __restrict__ P64, const float* __restrict__ P128,
                      float* __restrict__ U)
{
  long long idx = blockIdx.x * 256ll + threadIdx.x;
  long long total = (long long)Bm * 340 * Dm;
  if (idx >= total) return;
  int e = (int)(idx % Dm); int tm1 = (int)((idx / Dm) % 340); int b = (int)(idx / ((long long)Dm * 340));
  int t = tm1 + 1;
  int k, g, j; tok_decode(t, k, g, j);
  int gy = j / g, gx = j % g;
  int c = e >> 8, py = (e >> 4) & 15, px = e & 15;
  const float* src; int sp;
  switch (k){ case 1: src = P16;  sp = 16; break;
              case 2: src = P32;  sp = 32; break;
              case 3: src = P64;  sp = 64; break;
              default: src = P128; sp = 128; }
  int oy = gy * 16 + py, ox = gx * 16 + px;
  const float W75[4] = {-0.0234375f, 0.2265625f, 0.8671875f, -0.0703125f};
  const float W25[4] = {-0.0703125f, 0.8671875f, 0.2265625f, -0.0234375f};
  int my = oy >> 1, phy = oy & 1; const float* wy = phy ? W25 : W75; int y0 = my - 2 + phy;
  int mx = ox >> 1, phx = ox & 1; const float* wx = phx ? W25 : W75; int x0 = mx - 2 + phx;
  const float* sb = src + (long long)(b * 3 + c) * sp * sp;
  float acc = 0.f;
  for (int a = 0; a < 4; ++a){
    int yy = min(max(y0 + a, 0), sp - 1);
    float row = 0.f;
    for (int q = 0; q < 4; ++q){
      int xx = min(max(x0 + q, 0), sp - 1);
      row += wx[q] * sb[yy * sp + xx];
    }
    acc += wy[a] * row;
  }
  U[idx] = acc;
}

// ---------------- base embeddings (POS computed on the fly) ----------------
__global__ void embed_k(const float* __restrict__ start_tok, const float* __restrict__ scale_emb,
                        const float* __restrict__ class_emb, const float* __restrict__ wpe,
                        const int* __restrict__ labels, float* __restrict__ X)
{
  long long idx = blockIdx.x * 256ll + threadIdx.x;
  long long total = (long long)Bm * Lm * Dm;
  if (idx >= total) return;
  int d = (int)(idx % Dm); int t = (int)((idx / Dm) % Lm); int b = (int)(idx / ((long long)Dm * Lm));
  int k, g, j; tok_decode(t, k, g, j);
  int gy = j / g, gx = j % g;
  int s = 16 << k;
  float cy = (gy + 0.5f) * 16.f / (float)s;
  float cx = (gx + 0.5f) * 16.f / (float)s;
  int q = d % 192; int seg = d / 192;
  float omega = __expf(-(float)q * 0.04797052154089154f);   // ln(10000)/192
  float coord = (seg < 2) ? cx : cy;
  float ang = coord * omega;
  float pos = (seg == 0 || seg == 2) ? __sinf(ang) : __cosf(ang);
  float v = pos + scale_emb[k * Dm + d] + class_emb[(long long)labels[b] * Dm + d] + wpe[t * Dm + d];
  if (t == 0) v += start_tok[d];
  X[idx] = v;
}

// ---------------- deterministic two-stage loss reduction ----------------
__global__ __launch_bounds__(256)
void loss_part_k(const float* __restrict__ P, const float* __restrict__ T,
                 float* __restrict__ part, long long n)
{
  __shared__ float red[256];
  int t = threadIdx.x;
  float s = 0.f;
  for (long long i = blockIdx.x * 256ll + t; i < n; i += (long long)gridDim.x * 256){
    float d = P[i] - T[i]; s += d * d;
  }
  red[t] = s; __syncthreads();
  for (int o = 128; o > 0; o >>= 1){ if (t < o) red[t] += red[t + o]; __syncthreads(); }
  if (t == 0) part[blockIdx.x] = red[0];
}

__global__ __launch_bounds__(256)
void loss_final_k(const float* __restrict__ part, int nb, float* __restrict__ out, float inv)
{
  __shared__ float red[256];
  int t = threadIdx.x; float s = 0.f;
  for (int i = t; i < nb; i += 256) s += part[i];
  red[t] = s; __syncthreads();
  for (int o = 128; o > 0; o >>= 1){ if (t < o) red[t] += red[t + o]; __syncthreads(); }
  if (t == 0) out[0] = red[0] * inv;
}

// ---------------- host launcher ----------------
extern "C" void kernel_launch(void* const* d_in, const int* in_sizes, int n_in,
                              void* d_out, int out_size, void* d_ws, size_t ws_size,
                              hipStream_t stream)
{
  (void)in_sizes; (void)n_in; (void)out_size; (void)ws_size;
  const float* imgs      = (const float*)d_in[0];
  const float* patch_w   = (const float*)d_in[1];
  const float* patch_b   = (const float*)d_in[2];
  const float* scale_emb = (const float*)d_in[3];
  const float* class_emb = (const float*)d_in[4];
  const float* start_tok = (const float*)d_in[5];
  const float* wpe       = (const float*)d_in[6];
  const float* ln1_g     = (const float*)d_in[7];
  const float* ln1_b     = (const float*)d_in[8];
  const float* attn_w    = (const float*)d_in[9];
  const float* attn_b    = (const float*)d_in[10];
  const float* aproj_w   = (const float*)d_in[11];
  const float* aproj_b   = (const float*)d_in[12];
  const float* ln2_g     = (const float*)d_in[13];
  const float* ln2_b     = (const float*)d_in[14];
  const float* fc_w      = (const float*)d_in[15];
  const float* fc_b      = (const float*)d_in[16];
  const float* mproj_w   = (const float*)d_in[17];
  const float* mproj_b   = (const float*)d_in[18];
  const float* lnf_g     = (const float*)d_in[19];
  const float* lnf_b     = (const float*)d_in[20];
  const float* head_w    = (const float*)d_in[21];
  const float* head_b    = (const float*)d_in[22];
  const int*   labels    = (const int*)d_in[23];

  float* ws = (float*)d_ws;
  size_t off = 0;
  auto alloc = [&](size_t n){ float* p = ws + off; off += (n + 63) & ~(size_t)63; return p; };
  float* P128  = alloc((size_t)16 * 3 * 128 * 128);
  float* P64   = alloc((size_t)16 * 3 * 64 * 64);
  float* P32   = alloc((size_t)16 * 3 * 32 * 32);
  float* P16   = alloc((size_t)16 * 3 * 16 * 16);
  float* TGT   = alloc((size_t)16 * 341 * 768);
  float* UPP   = alloc((size_t)16 * 340 * 768);
  float* X     = alloc((size_t)16 * 341 * 768);
  float* H     = alloc((size_t)16 * 341 * 768);
  float* QKV   = alloc((size_t)16 * 341 * 2304);       // also reused as PRED
  float* OBUF  = alloc((size_t)16 * 341 * 768);
  float* SC    = alloc((size_t)16 * 12 * 341 * SLD);   // scores; MLP hidden aliases this
  float* PARTS = alloc((size_t)1024);
  float* MLP   = SC;                                   // alias: scores done before MLP starts
  float* PRED  = QKV;

  dim3 blk(128);
  auto gemm = [&](const float* A, int lda, long long asb, long long ash,
                  const float* B, int ldb, long long bsb, long long bsh, int bnk,
                  float* C, int ldc, long long csb, long long csh,
                  const float* bias, const float* res,
                  int M, int N, int K, int Z, int zh, int flags){
    dim3 grd((unsigned)((N + 63) / 64), (unsigned)((M + 127) / 128), (unsigned)Z);
    gemm_wmma<<<grd, blk, 0, stream>>>(A, lda, asb, ash, B, ldb, bsb, bsh, bnk,
                                       C, ldc, csb, csh, bias, res, M, N, K, zh, flags);
  };

  // ---- preprocessing ----
  { int n = 16*3*128*128; pool2x2_k<<<(n+255)/256, 256, 0, stream>>>(imgs, P128, n, 128); }
  { int n = 16*3*64*64;   pool2x2_k<<<(n+255)/256, 256, 0, stream>>>(P128, P64,  n, 64); }
  { int n = 16*3*32*32;   pool2x2_k<<<(n+255)/256, 256, 0, stream>>>(P64,  P32,  n, 32); }
  { int n = 16*3*16*16;   pool2x2_k<<<(n+255)/256, 256, 0, stream>>>(P32,  P16,  n, 16); }
  { long long n = 16ll*341*768; targets_k<<<(unsigned)((n+255)/256), 256, 0, stream>>>(imgs, P16, P32, P64, P128, TGT); }
  { long long n = 16ll*340*768; upp_k<<<(unsigned)((n+255)/256), 256, 0, stream>>>(P16, P32, P64, P128, UPP); }
  { long long n = 16ll*341*768; embed_k<<<(unsigned)((n+255)/256), 256, 0, stream>>>(start_tok, scale_emb, class_emb, wpe, labels, X); }
  // patch-embed of upsampled patches, accumulated into x (tokens 1..340 per image)
  gemm(UPP, 768, 0, 0, patch_w, 768, 0, 0, 0, X, 768, 0, 0, patch_b, X,
       16*340, 768, 768, 1, 1, GF_BIAS | GF_RES | GF_REMAP);

  const int ROWS = 16 * 341;               // 5456
  // ---- transformer layers ----
  for (int i = 0; i < 12; ++i){
    layernorm_k<<<ROWS, 256, 0, stream>>>(X, ln1_g + i*768, ln1_b + i*768, H);
    gemm(H, 768, 0, 0, attn_w + (long long)i*768*2304, 2304, 0, 0, 0,
         QKV, 2304, 0, 0, attn_b + i*2304, nullptr, ROWS, 2304, 768, 1, 1, GF_BIAS);
    // scores = q . k^T  (per b,h)
    gemm(QKV, 2304, 341ll*2304, 64, QKV + 768, 2304, 341ll*2304, 64, 1,
         SC, SLD, 12ll*341*SLD, 341ll*SLD, nullptr, nullptr,
         341, 341, 64, 16*12, 12, 0);
    softmax_mask_k<<<16*12*341, 128, 0, stream>>>(SC);
    // o = P . V
    gemm(SC, SLD, 12ll*341*SLD, 341ll*SLD, QKV + 1536, 2304, 341ll*2304, 64, 0,
         OBUF, 768, 341ll*768, 64, nullptr, nullptr,
         341, 64, 341, 16*12, 12, 0);
    gemm(OBUF, 768, 0, 0, aproj_w + (long long)i*768*768, 768, 0, 0, 0,
         X, 768, 0, 0, aproj_b + i*768, X, ROWS, 768, 768, 1, 1, GF_BIAS | GF_RES);
    layernorm_k<<<ROWS, 256, 0, stream>>>(X, ln2_g + i*768, ln2_b + i*768, H);
    gemm(H, 768, 0, 0, fc_w + (long long)i*768*3072, 3072, 0, 0, 0,
         MLP, 3072, 0, 0, fc_b + i*3072, nullptr, ROWS, 3072, 768, 1, 1, GF_BIAS | GF_GELU);
    gemm(MLP, 3072, 0, 0, mproj_w + (long long)i*3072*768, 768, 0, 0, 0,
         X, 768, 0, 0, mproj_b + i*768, X, ROWS, 768, 3072, 1, 1, GF_BIAS | GF_RES);
  }

  // ---- head + loss ----
  layernorm_k<<<ROWS, 256, 0, stream>>>(X, lnf_g, lnf_b, H);
  gemm(H, 768, 0, 0, head_w, 768, 0, 0, 0, PRED, 768, 0, 0,
       head_b, nullptr, ROWS, 768, 768, 1, 1, GF_BIAS);
  long long n = 16ll * 341 * 768;
  loss_part_k<<<1024, 256, 0, stream>>>(PRED, TGT, PARTS, n);
  loss_final_k<<<1, 256, 0, stream>>>(PARTS, 1024, (float*)d_out, 1.f / (float)n);
}
